// PointTransformerV2MambaVoxel_58789512348497
// MI455X (gfx1250) — matrix operand order repaired and checked by reading, not compiled, not run on hardware
//
#include <hip/hip_runtime.h>
#include <hip/hip_bf16.h>
#include <stdint.h>

// ---------------------------------------------------------------------------
// MI455X (gfx1250) implementation of PointTransformerV2-Mamba block.
//  * GEMMs via v_wmma_f32_16x16x32_f16 (f16 operands, f32 accumulate).
//  * Weights pre-packed into WMMA per-lane B layout -> 2x global_load_b128.
//  * A tiles DMA'd global->LDS with global_load_async_to_lds_b128 (ASYNCcnt),
//    then converted f32->f16 in LDS; operands read as 2x ds_load_b128.
//  * Fused epilogues: bias / LayerNorm / ReLU / softplus / residual.
//  * Selective scan as 3-pass chunked parallel scan over 4096 independent
//    (d,s) recurrences; y-reduction via width-16 shfl_xor (wave32).
// ---------------------------------------------------------------------------

typedef _Float16 v16h __attribute__((ext_vector_type(16)));
typedef _Float16 v8h  __attribute__((ext_vector_type(8)));
typedef float    v8f  __attribute__((ext_vector_type(8)));
typedef __attribute__((address_space(3))) float lds_f32;

#define NPTS   16384
#define CDIM   128
#define DIN    256      // D_INNER
#define DST    16       // D_STATE
#define DTRK   8        // DT_RANK
#define XDBL_W 48       // 8 + 16 + 16, padded to 48 (multiple of 16)
#define EPSLN  1e-5f

__device__ __forceinline__ float siluf(float x)     { return x / (1.f + __expf(-x)); }
__device__ __forceinline__ float softplusf(float x) { return (x > 20.f) ? x : log1pf(__expf(x)); }

// 128-thread block sum reduction
__device__ __forceinline__ float reduce128(float v, float* s) {
    int t = threadIdx.x;
    s[t] = v; __syncthreads();
    #pragma unroll
    for (int off = 64; off > 0; off >>= 1) {
        if (t < off) s[t] += s[t + off];
        __syncthreads();
    }
    float r = s[0]; __syncthreads();
    return r;
}

// ---------------------------------------------------------------------------
// Weight packer: B[k,n] (optionally W transposed), zero-padded, laid out so
// that lane `lane` of (ktile kt, ntile nt) holds 16 contiguous halfwords:
//   element i  <->  k = kt*32 + 16*(lane>>4) + i,  n = nt*16 + (lane&15)
// ---------------------------------------------------------------------------
__global__ void k_pack_b(const float* __restrict__ W, _Float16* __restrict__ out,
                         int ldw, int kreal, int nreal, int transpose)
{
    int kt = blockIdx.x, nt = blockIdx.y, ntiles = gridDim.y;
    int lane = threadIdx.x;          // 0..31
    int n = nt * 16 + (lane & 15);
    int kb = kt * 32 + 16 * (lane >> 4);
    _Float16* dst = out + (((size_t)(kt * ntiles + nt)) * 32 + lane) * 16;
    #pragma unroll
    for (int i = 0; i < 16; ++i) {
        int k = kb + i;
        float v = 0.f;
        if (k < kreal && n < nreal)
            v = transpose ? W[(size_t)n * ldw + k] : W[(size_t)k * ldw + n];
        dst[i] = (_Float16)v;
    }
}

// ---------------------------------------------------------------------------
// WMMA GEMM: Out[16384 x ncols] = A[16384 x kreal] * Bpacked[KP x ncols]
// EPI: 0 = (+bias) raw store, 1 = (+bias) softplus store,
//      2 = (+bias) LayerNorm(gamma,beta) (+residual) ReLU   [ncols must be 128]
// ASYNCA: stage A tile via global_load_async_to_lds_b128 (requires kreal==KP,
//         16B-aligned rows); otherwise VGPR staging with zero-pad.
// Block: 256 threads = 8 waves; block owns 16 rows; wave w owns n-tiles
// w, w+8, ...  Grid.x = 16384/16 = 1024.
// ---------------------------------------------------------------------------
template<int KP, int EPI, bool HAS_BIAS, bool HAS_RES, bool ASYNCA>
__global__ __launch_bounds__(256)
void k_gemm(const float* __restrict__ A, int lda, int kreal,
            const _Float16* __restrict__ Bp, int ntiles,
            const float* __restrict__ bias,
            const float* __restrict__ gamma, const float* __restrict__ beta,
            const float* __restrict__ resid,
            float* __restrict__ Out, int ldo)
{
    constexpr int KT    = KP / 32;
    constexpr int ECOLS = (EPI == 2) ? 132 : 1;
    __shared__ __align__(16) _Float16 As[16][KP + 8];
    __shared__ __align__(16) float Asf[ASYNCA ? 16 : 1][ASYNCA ? (KP + 4) : 1];
    __shared__ float Es[(EPI == 2) ? 16 : 1][ECOLS];
    __shared__ float muS[16], rsS[16];

    const int tid   = threadIdx.x;
    const int mbase = blockIdx.x * 16;

    if constexpr (ASYNCA) {
        // DMA raw f32 A tile into LDS (no VGPR round trip), 16B per lane-op.
        constexpr int CPR = KP / 4;                 // 16B chunks per row
        for (int ci = tid; ci < 16 * CPR; ci += 256) {
            int row = ci / CPR, c4 = ci % CPR;
            uint64_t gaddr = (uint64_t)(uintptr_t)(A + (size_t)(mbase + row) * lda + c4 * 4);
            uint32_t loff  = (uint32_t)(uintptr_t)(lds_f32*)&Asf[row][c4 * 4];
            asm volatile("global_load_async_to_lds_b128 %0, %1, off"
                         :: "v"(loff), "v"(gaddr) : "memory");
        }
        asm volatile("s_wait_asynccnt 0x0" ::: "memory");
        __syncthreads();
        // f32 -> f16 conversion pass (LDS -> LDS)
        for (int idx = tid; idx < 16 * KP; idx += 256) {
            int row = idx / KP, col = idx % KP;
            As[row][col] = (_Float16)Asf[row][col];
        }
        __syncthreads();
    } else {
        // VGPR staging with zero-pad for k >= kreal.
        for (int idx = tid; idx < 16 * KP; idx += 256) {
            int row = idx / KP, col = idx % KP;
            float v = (col < kreal) ? A[(size_t)(mbase + row) * lda + col] : 0.f;
            As[row][col] = (_Float16)v;
        }
        __syncthreads();
    }

    const int lane = tid & 31;
    const int wave = tid >> 5;
    const int half = lane >> 4;
    const int mloc = lane & 15;

    for (int nt = wave; nt < ntiles; nt += 8) {
        v8f acc = {};
        #pragma unroll
        for (int kt = 0; kt < KT; ++kt) {
            // A operand: element i -> k = kt*32 + 16*half + i
            const _Float16* arow = &As[mloc][kt * 32 + 16 * half];
            v8h alo = *(const v8h*)arow;
            v8h ahi = *(const v8h*)(arow + 8);
            v16h av = __builtin_shufflevector(alo, ahi,
                        0,1,2,3,4,5,6,7,8,9,10,11,12,13,14,15);
            // B operand: contiguous 16 halfwords per lane (pre-swizzled)
            const _Float16* bptr = Bp + (((size_t)(kt * ntiles + nt)) * 32 + lane) * 16;
            if (kt + 1 < KT)
                __builtin_prefetch(bptr + (size_t)ntiles * 512, 0, 3);
            v8h blo = *(const v8h*)bptr;
            v8h bhi = *(const v8h*)(bptr + 8);
            v16h bv = __builtin_shufflevector(blo, bhi,
                        0,1,2,3,4,5,6,7,8,9,10,11,12,13,14,15);
            acc = __builtin_amdgcn_wmma_f32_16x16x32_f16(
                      false, av, false, bv, (short)0, acc, false, false);
        }
        if constexpr (EPI == 2) {
            #pragma unroll
            for (int r = 0; r < 8; ++r) {
                int row = r + 8 * half;
                int col = nt * 16 + mloc;
                float v = acc[r];
                if constexpr (HAS_BIAS) v += bias[col];
                Es[row][col] = v;
            }
        } else {
            #pragma unroll
            for (int r = 0; r < 8; ++r) {
                int row = mbase + r + 8 * half;
                int col = nt * 16 + mloc;
                float v = acc[r];
                if constexpr (HAS_BIAS) v += bias[col];
                if constexpr (EPI == 1) v = softplusf(v);
                Out[(size_t)row * ldo + col] = v;
            }
        }
    }

    if constexpr (EPI == 2) {
        __syncthreads();
        if (tid < 16) {
            float s1 = 0.f, s2 = 0.f;
            for (int c = 0; c < 128; ++c) {
                float v = Es[tid][c];
                s1 += v; s2 += v * v;
            }
            float mu  = s1 * (1.f / 128.f);
            float var = s2 * (1.f / 128.f) - mu * mu;
            muS[tid] = mu;
            rsS[tid] = rsqrtf(var + EPSLN);
        }
        __syncthreads();
        for (int idx = tid; idx < 16 * 128; idx += 256) {
            int row = idx >> 7, col = idx & 127;
            float v = (Es[row][col] - muS[row]) * rsS[row] * gamma[col] + beta[col];
            if constexpr (HAS_RES) v += resid[(size_t)(mbase + row) * 128 + col];
            v = fmaxf(v, 0.f);
            Out[(size_t)(mbase + row) * ldo + col] = v;
        }
    }
}

// ---------------------------------------------------------------------------
// Gather + gaussian-weighted sum + LayerNorm -> cat[:,128:256]
// ---------------------------------------------------------------------------
__global__ __launch_bounds__(128)
void k_gather_ln(const float* __restrict__ f, const int* __restrict__ refi,
                 const float* __restrict__ gd,
                 const float* __restrict__ g, const float* __restrict__ b,
                 float* __restrict__ cat)
{
    __shared__ float red[128];
    int n = blockIdx.x, c = threadIdx.x;
    float acc = 0.f;
    #pragma unroll 4
    for (int k = 0; k < 16; ++k) {
        int idx = refi[n * 16 + k];
        acc += gd[n * 16 + k] * f[(size_t)idx * 128 + c];
    }
    float mu = reduce128(acc, red) * (1.f / 128.f);
    float dv = acc - mu;
    float var = reduce128(dv * dv, red) * (1.f / 128.f);
    cat[(size_t)n * 256 + 128 + c] = dv * rsqrtf(var + EPSLN) * g[c] + b[c];
}

// feat_mamba = LN(f + m_out) -> cat[:,0:128]
__global__ __launch_bounds__(128)
void k_add_ln(const float* __restrict__ f, const float* __restrict__ mo,
              const float* __restrict__ g, const float* __restrict__ b,
              float* __restrict__ cat)
{
    __shared__ float red[128];
    int n = blockIdx.x, c = threadIdx.x;
    float v = f[(size_t)n * 128 + c] + mo[(size_t)n * 128 + c];
    float mu = reduce128(v, red) * (1.f / 128.f);
    float dv = v - mu;
    float var = reduce128(dv * dv, red) * (1.f / 128.f);
    cat[(size_t)n * 256 + c] = dv * rsqrtf(var + EPSLN) * g[c] + b[c];
}

// xr = rmsnorm(f, w)
__global__ __launch_bounds__(128)
void k_rmsnorm(const float* __restrict__ f, const float* __restrict__ w,
               float* __restrict__ xr)
{
    __shared__ float red[128];
    int n = blockIdx.x, c = threadIdx.x;
    float v  = f[(size_t)n * 128 + c];
    float ms = reduce128(v * v, red) * (1.f / 128.f);
    xr[(size_t)n * 128 + c] = v * rsqrtf(ms + EPSLN) * w[c];
}

// Causal depthwise conv (k=4) + bias + silu:  xm[n,d]
__global__ __launch_bounds__(256)
void k_conv_silu(const float* __restrict__ xz, const float* __restrict__ cw,
                 const float* __restrict__ cb, float* __restrict__ xm)
{
    int i = blockIdx.x * 256 + threadIdx.x;   // i = n*256 + d
    int n = i >> 8, d = i & 255;
    float acc = cb[d];
    #pragma unroll
    for (int j = 0; j < 4; ++j) {
        int t = n - 3 + j;
        if (t >= 0) acc += cw[d * 4 + j] * xz[(size_t)t * 512 + d];
    }
    xm[i] = siluf(acc);
}

// y = (y + D*xm) * silu(z)   (in place)
__global__ __launch_bounds__(256)
void k_gate(float* __restrict__ y, const float* __restrict__ xm,
            const float* __restrict__ Dp, const float* __restrict__ xz)
{
    int i = blockIdx.x * 256 + threadIdx.x;
    int n = i >> 8, d = i & 255;
    float v = y[i] + Dp[d] * xm[i];
    float z = xz[(size_t)n * 512 + 256 + d];
    y[i] = v * siluf(z);
}

// ---------------------------------------------------------------------------
// Chunked parallel selective scan. 128 chunks x 128 steps.
// ---------------------------------------------------------------------------
__global__ __launch_bounds__(256)
void k_scan_part1(const float* __restrict__ delta, const float* __restrict__ xm,
                  const float* __restrict__ xdbl, const float* __restrict__ A_log,
                  float* __restrict__ chunkA, float* __restrict__ chunkB)
{
    int tid = threadIdx.x;
    int s = tid & 15, dl = tid >> 4;
    int c = blockIdx.x;
    int d = blockIdx.y * 16 + dl;
    float Ads = -__expf(A_log[d * 16 + s]);
    float arun = 1.f, h = 0.f;
    int t0 = c * 128;
    for (int tt = 0; tt < 128; ++tt) {
        int t = t0 + tt;
        float dlt = delta[(size_t)t * 256 + d];
        float xv  = xm  [(size_t)t * 256 + d];
        float Bv  = xdbl[(size_t)t * XDBL_W + DTRK + s];
        float a = __expf(dlt * Ads);
        arun *= a;
        h = a * h + dlt * Bv * xv;
    }
    size_t o = ((size_t)c * 256 + d) * 16 + s;
    chunkA[o] = arun;
    chunkB[o] = h;
}

__global__ __launch_bounds__(1024)
void k_scan_part2(const float* __restrict__ chunkA, const float* __restrict__ chunkB,
                  float* __restrict__ carry)
{
    for (int pair = threadIdx.x; pair < 4096; pair += 1024) {
        float h = 0.f;
        for (int c = 0; c < 128; ++c) {
            size_t o = (size_t)c * 4096 + pair;
            carry[o] = h;
            h = chunkA[o] * h + chunkB[o];
        }
    }
}

__global__ __launch_bounds__(256)
void k_scan_part3(const float* __restrict__ delta, const float* __restrict__ xm,
                  const float* __restrict__ xdbl, const float* __restrict__ A_log,
                  const float* __restrict__ carry, float* __restrict__ y)
{
    int tid = threadIdx.x;
    int s = tid & 15, dl = tid >> 4;
    int c = blockIdx.x;
    int d = blockIdx.y * 16 + dl;
    float Ads = -__expf(A_log[d * 16 + s]);
    float h = carry[(size_t)c * 4096 + (size_t)(d * 16 + s)];
    int t0 = c * 128;
    for (int tt = 0; tt < 128; ++tt) {
        int t = t0 + tt;
        float dlt = delta[(size_t)t * 256 + d];
        float xv  = xm  [(size_t)t * 256 + d];
        float Bv  = xdbl[(size_t)t * XDBL_W + DTRK + s];
        float Cv  = xdbl[(size_t)t * XDBL_W + DTRK + DST + s];
        float a = __expf(dlt * Ads);
        h = a * h + dlt * Bv * xv;
        float p = h * Cv;
        p += __shfl_xor(p, 1, 16);
        p += __shfl_xor(p, 2, 16);
        p += __shfl_xor(p, 4, 16);
        p += __shfl_xor(p, 8, 16);
        if (s == 0) y[(size_t)t * 256 + d] = p;
    }
}

// ---------------------------------------------------------------------------
// Host orchestration
// ---------------------------------------------------------------------------
extern "C" void kernel_launch(void* const* d_in, const int* in_sizes, int n_in,
                              void* d_out, int out_size, void* d_ws, size_t ws_size,
                              hipStream_t stream)
{
    (void)in_sizes; (void)n_in; (void)out_size; (void)ws_size;

    const float* feat   = (const float*)d_in[0];
    const int*   refi   = (const int*)  d_in[1];
    const float* gauss  = (const float*)d_in[2];
    const float* fc1_w  = (const float*)d_in[3];
    const float* fc3_w  = (const float*)d_in[4];
    const float* n1g    = (const float*)d_in[5];
    const float* n1b    = (const float*)d_in[6];
    const float* n2g    = (const float*)d_in[7];
    const float* n2b    = (const float*)d_in[8];
    const float* n3g    = (const float*)d_in[9];
    const float* n3b    = (const float*)d_in[10];
    const float* gvg    = (const float*)d_in[11];
    const float* gvb    = (const float*)d_in[12];
    const float* la_w1  = (const float*)d_in[13];
    const float* la_b1  = (const float*)d_in[14];
    const float* lng    = (const float*)d_in[15];
    const float* lnb    = (const float*)d_in[16];
    const float* la_w2  = (const float*)d_in[17];
    const float* la_b2  = (const float*)d_in[18];
    const float* rmsw   = (const float*)d_in[19];
    const float* in_w   = (const float*)d_in[20];
    const float* conv_w = (const float*)d_in[21];
    const float* conv_b = (const float*)d_in[22];
    const float* xp_w   = (const float*)d_in[23];
    const float* dt_w   = (const float*)d_in[24];
    const float* dt_b   = (const float*)d_in[25];
    const float* A_log  = (const float*)d_in[26];
    const float* Dp     = (const float*)d_in[27];
    const float* out_w  = (const float*)d_in[28];
    float* out = (float*)d_out;

    // ---- workspace layout ----
    _Float16* wp = (_Float16*)d_ws;
    _Float16* w_fc1 = wp;            // 128x128  -> 16384
    _Float16* w_in  = wp + 16384;    // 128x512  -> 65536
    _Float16* w_xp  = wp + 81920;    // 256x48   -> 12288
    _Float16* w_dt  = wp + 94208;    // 32x256   -> 8192
    _Float16* w_out = wp + 102400;   // 256x128  -> 32768
    _Float16* w_la1 = wp + 135168;   // 256x128  -> 32768
    _Float16* w_la2 = wp + 167936;   // 128x128  -> 16384
    _Float16* w_fc3 = wp + 184320;   // 128x128  -> 16384 (ends 200704 f16)

    float* fb    = (float*)((char*)d_ws + 401408);   // 16B aligned
    float* f     = fb;               // 16384x128
    float* xr    = fb + 2097152;     // 16384x128
    float* xz    = fb + 4194304;     // 16384x512
    float* xm    = fb + 12582912;    // 16384x256
    float* xdbl  = fb + 16777216;    // 16384x48
    float* delta = fb + 17563648;    // 16384x256
    float* y     = fb + 21757952;    // 16384x256
    float* mout  = fb + 25952256;    // 16384x128
    float* cat   = fb + 28049408;    // 16384x256
    float* res   = fb + 32243712;    // 16384x128
    float* f2    = fb + 34340864;    // 16384x128
    float* cA    = fb + 36438016;    // 128x256x16
    float* cB    = fb + 36962304;    // 128x256x16
    float* carry = fb + 37486592;    // 128x256x16

    // ---- pack weights into WMMA B layout (f16) ----
    k_pack_b<<<dim3(4,  8), 32, 0, stream>>>(fc1_w, w_fc1, 128, 128, 128, 1);
    k_pack_b<<<dim3(4, 32), 32, 0, stream>>>(in_w,  w_in,  512, 128, 512, 0);
    k_pack_b<<<dim3(8,  3), 32, 0, stream>>>(xp_w,  w_xp,   40, 256,  40, 0);
    k_pack_b<<<dim3(1, 16), 32, 0, stream>>>(dt_w,  w_dt,  256,   8, 256, 0);
    k_pack_b<<<dim3(8,  8), 32, 0, stream>>>(out_w, w_out, 128, 256, 128, 0);
    k_pack_b<<<dim3(8,  8), 32, 0, stream>>>(la_w1, w_la1, 128, 256, 128, 0);
    k_pack_b<<<dim3(4,  8), 32, 0, stream>>>(la_w2, w_la2, 128, 128, 128, 0);
    k_pack_b<<<dim3(4,  8), 32, 0, stream>>>(fc3_w, w_fc3, 128, 128, 128, 1);

    const int GB = NPTS / 16;          // 1024 GEMM row-blocks
    const int EW = NPTS * 256 / 256;   // elementwise grid over N*256

    // f = relu(LN(feat @ fc1_w.T, norm1))
    k_gemm<128, 2, false, false, true><<<GB, 256, 0, stream>>>(
        feat, 128, 128, w_fc1, 8, nullptr, n1g, n1b, nullptr, f, 128);

    // combined -> cat[:,128:]
    k_gather_ln<<<NPTS, 128, 0, stream>>>(f, refi, gauss, gvg, gvb, cat);

    // xr = rmsnorm(f);  xz = xr @ m_in_w
    k_rmsnorm<<<NPTS, 128, 0, stream>>>(f, rmsw, xr);
    k_gemm<128, 0, false, false, true><<<GB, 256, 0, stream>>>(
        xr, 128, 128, w_in, 32, nullptr, nullptr, nullptr, nullptr, xz, 512);

    // xm = silu(conv1d(xz[:, :256]) + b)
    k_conv_silu<<<EW, 256, 0, stream>>>(xz, conv_w, conv_b, xm);

    // x_dbl = xm @ m_xproj_w  (N padded 40 -> 48)
    k_gemm<256, 0, false, false, true><<<GB, 256, 0, stream>>>(
        xm, 256, 256, w_xp, 3, nullptr, nullptr, nullptr, nullptr, xdbl, XDBL_W);

    // delta = softplus(x_dbl[:, :8] @ m_dt_w + m_dt_b)   (K padded 8 -> 32)
    k_gemm<32, 1, true, false, false><<<GB, 256, 0, stream>>>(
        xdbl, XDBL_W, DTRK, w_dt, 16, dt_b, nullptr, nullptr, nullptr, delta, 256);

    // chunked selective scan -> y
    k_scan_part1<<<dim3(128, 16), 256, 0, stream>>>(delta, xm, xdbl, A_log, cA, cB);
    k_scan_part2<<<1, 1024, 0, stream>>>(cA, cB, carry);
    k_scan_part3<<<dim3(128, 16), 256, 0, stream>>>(delta, xm, xdbl, A_log, carry, y);

    // y = (y + D*xm) * silu(z);   mout = y @ m_out_w
    k_gate<<<EW, 256, 0, stream>>>(y, xm, Dp, xz);
    k_gemm<256, 0, false, false, true><<<GB, 256, 0, stream>>>(
        y, 256, 256, w_out, 8, nullptr, nullptr, nullptr, nullptr, mout, 128);

    // feat_mamba = LN(f + mout) -> cat[:, :128]
    k_add_ln<<<NPTS, 128, 0, stream>>>(f, mout, gvg, gvb, cat);

    // res = relu(LN(cat @ la_w1 + b1, la_norm))
    k_gemm<256, 2, true, false, true><<<GB, 256, 0, stream>>>(
        cat, 256, 256, w_la1, 8, la_b1, lng, lnb, nullptr, res, 128);

    // f2 = relu(LN(res @ la_w2 + b2, norm2))
    k_gemm<128, 2, true, false, true><<<GB, 256, 0, stream>>>(
        res, 128, 128, w_la2, 8, la_b2, n2g, n2b, nullptr, f2, 128);

    // out = relu(feat + LN(f2 @ fc3_w.T, norm3))
    k_gemm<128, 2, false, true, true><<<GB, 256, 0, stream>>>(
        f2, 128, 128, w_fc3, 8, nullptr, n3g, n3b, feat, out, 128);
}
